// OwnMultiHeadTransformer_33277406609761
// MI455X (gfx1250) — matrix operand order repaired and checked by reading, not compiled, not run on hardware
//
#include <hip/hip_runtime.h>
#include <cstdint>
#include <cstddef>

// ---------------------------------------------------------------------------
// MI455X (gfx1250, wave32) transformer encoder layer.
// All matmuls through v_wmma_f32_16x16x32_bf16 (f32 accumulate).
// Tile staging uses GLOBAL_LOAD_ASYNC_TO_LDS (ASYNCcnt) when available.
// ---------------------------------------------------------------------------

typedef __bf16 bf16_t;
typedef __attribute__((ext_vector_type(16))) __bf16 v16bf;
typedef __attribute__((ext_vector_type(8)))  float  v8f;

constexpr int kB  = 4;
constexpr int kS  = 2048;
constexpr int kD  = 512;
constexpr int kH  = 8;
constexpr int kFF = 2048;
constexpr int kHD = 64;
constexpr int kM  = kB * kS;            // 8192 rows total
constexpr float kInvScale = 0.04419417382415922f;   // 1/sqrt(512)

#if __has_builtin(__builtin_amdgcn_global_load_async_to_lds_b128) && \
    __has_builtin(__builtin_amdgcn_s_wait_asynccnt)
#define HAVE_ASYNC_LDS 1
#else
#define HAVE_ASYNC_LDS 0
#endif

#define AS_GLOBAL __attribute__((address_space(1)))
#define AS_LOCAL  __attribute__((address_space(3)))

// GCC-style int4 vector: the async-LDS builtins take pointers to this type.
typedef int gv4i __attribute__((vector_size(16)));

#if HAVE_ASYNC_LDS
// 16B global -> LDS direct copy, tracked by ASYNCcnt (no VGPR data path).
__device__ __forceinline__ void async_copy_b128(void* lds_dst, const void* gsrc) {
  __builtin_amdgcn_global_load_async_to_lds_b128(
      (AS_GLOBAL gv4i*)gsrc, (AS_LOCAL gv4i*)lds_dst, 0, 0);
}
__device__ __forceinline__ void async_wait0() {
  __builtin_amdgcn_s_wait_asynccnt(0);
}
#endif

// ---- WMMA helpers ----------------------------------------------------------

__device__ __forceinline__ v8f wmma_bf16(v16bf a, v16bf b, v8f c) {
  // 8 args: (neg_a, A, neg_b, B, c_mod, C, reuse_a, reuse_b)
  return __builtin_amdgcn_wmma_f32_16x16x32_bf16(false, a, false, b, (short)0, c,
                                                 false, false);
}

union FragU {
  uint4 u[2];
  v16bf v;
};

// A-operand fragment (16x32, M x K): caller passes p = &A[row(lane)][kb + hi*8]
// elems 0..7 = p[0..7] (K lo half), elems 8..15 = p[16..23] (K hi half).
__device__ __forceinline__ v16bf fragA_ld(const bf16_t* p) {
  FragU f;
  f.u[0] = *reinterpret_cast<const uint4*>(p);
  f.u[1] = *reinterpret_cast<const uint4*>(p + 16);
  return f.v;
}

// B-operand fragment (32x16, K x N) from LDS with row stride 72: lane = N col,
// caller passes p = &B[kb + hi*8][nb + lr]; elems 0..7 = K rows 0..7, 8..15 = K rows 16..23.
__device__ __forceinline__ v16bf fragB_ld(const bf16_t* p) {
  v16bf v;
#pragma unroll
  for (int s = 0; s < 8; ++s) {
    v[s] = p[s * 72];
    v[s + 8] = p[(s + 16) * 72];
  }
  return v;
}

__device__ __forceinline__ v8f v8f_zero() {
  v8f z;
#pragma unroll
  for (int i = 0; i < 8; ++i) z[i] = 0.0f;
  return z;
}

// ---- f32 -> bf16 convert ---------------------------------------------------

__global__ void cvt_kernel(const float* __restrict__ in, bf16_t* __restrict__ out, int n) {
  int i = blockIdx.x * 256 + threadIdx.x;
  if (i < n) out[i] = (bf16_t)in[i];
}

// ---- Generic bf16 GEMM: C[m][n] = (sum_k A[m][k]*Bblk[k][n] + bias[n]) -----
// B is addressed as: Bmat + (n_block/64)*nbstride + k*bld + (n%64-tile col)
//   weights [K][N]:      bld = N,  nbstride = 64       (plain row-major)
//   per-head QKV [H][D][HD]: bld = 64, nbstride = D*HD (one head per n-block)
// MODE 0: (+bias)*cscale -> bf16 (QKV; Q folds 1/sqrt(D) into cscale)
// MODE 1: +bias +resid -> f32    (attn out proj + residual)
// MODE 2: +bias, relu -> bf16    (FFN1)
// MODE 3: +bias -> f32           (FFN2 -> d_out)

template <int MODE>
__global__ __launch_bounds__(256) void gemm_bf16_kernel(
    const bf16_t* __restrict__ A, const bf16_t* __restrict__ Bmat,
    const float* __restrict__ bias, const float* __restrict__ resid,
    bf16_t* __restrict__ outb, float* __restrict__ outf,
    int M, int N, int K, int bld, long nbstride, float cscale) {
  __shared__ bf16_t As[128][72];
  __shared__ bf16_t Bs[64][72];

  const int tid  = threadIdx.x;
  const int lane = tid & 31;
  const int wid  = tid >> 5;
  const int lr   = lane & 15;
  const int hi   = lane >> 4;
  const int wr   = wid >> 1;       // 0..3
  const int wc   = wid & 1;        // 0..1
  const int mb   = blockIdx.y * 128;
  const int nb   = blockIdx.x * 64;
  const int m0   = wr * 32;
  const int n0   = wc * 32;

  const bf16_t* Bblk = Bmat + (size_t)(nb >> 6) * (size_t)nbstride;

  v8f acc[2][2];
#pragma unroll
  for (int i = 0; i < 2; ++i)
#pragma unroll
    for (int j = 0; j < 2; ++j) acc[i][j] = v8f_zero();

  const int arow = tid >> 1, acol = (tid & 1) * 32;   // 128 rows x 64 cols
  const int brow = tid >> 2, bcol = (tid & 3) * 16;   // 64 rows x 64 cols

  for (int k0 = 0; k0 < K; k0 += 64) {
    __syncthreads();
    const bf16_t* srcA = A + (size_t)(mb + arow) * K + k0 + acol;
    const bf16_t* srcB = Bblk + (size_t)(k0 + brow) * bld + bcol;
#if HAVE_ASYNC_LDS
    async_copy_b128(&As[arow][acol +  0], srcA +  0);
    async_copy_b128(&As[arow][acol +  8], srcA +  8);
    async_copy_b128(&As[arow][acol + 16], srcA + 16);
    async_copy_b128(&As[arow][acol + 24], srcA + 24);
    async_copy_b128(&Bs[brow][bcol + 0], srcB + 0);
    async_copy_b128(&Bs[brow][bcol + 8], srcB + 8);
#else
    {
      const uint4* s4 = reinterpret_cast<const uint4*>(srcA);
      uint4* d4 = reinterpret_cast<uint4*>(&As[arow][acol]);
      d4[0] = s4[0]; d4[1] = s4[1]; d4[2] = s4[2]; d4[3] = s4[3];
    }
    {
      const uint4* s4 = reinterpret_cast<const uint4*>(srcB);
      uint4* d4 = reinterpret_cast<uint4*>(&Bs[brow][bcol]);
      d4[0] = s4[0]; d4[1] = s4[1];
    }
#endif
    if (k0 + 64 < K) {  // global_prefetch_b8 of next A tile
      __builtin_prefetch(A + (size_t)(mb + arow) * K + k0 + 64 + acol, 0, 1);
    }
#if HAVE_ASYNC_LDS
    async_wait0();
#endif
    __syncthreads();

#pragma unroll
    for (int ks = 0; ks < 64; ks += 32) {
      v16bf a0 = fragA_ld(&As[m0 + lr][ks + hi * 8]);
      v16bf a1 = fragA_ld(&As[m0 + 16 + lr][ks + hi * 8]);
      v16bf b0 = fragB_ld(&Bs[ks + hi * 8][n0 + lr]);
      v16bf b1 = fragB_ld(&Bs[ks + hi * 8][n0 + 16 + lr]);
      acc[0][0] = wmma_bf16(a0, b0, acc[0][0]);
      acc[0][1] = wmma_bf16(a0, b1, acc[0][1]);
      acc[1][0] = wmma_bf16(a1, b0, acc[1][0]);
      acc[1][1] = wmma_bf16(a1, b1, acc[1][1]);
    }
  }

#pragma unroll
  for (int ti = 0; ti < 2; ++ti)
#pragma unroll
    for (int tj = 0; tj < 2; ++tj)
#pragma unroll
      for (int j = 0; j < 8; ++j) {
        int m = mb + m0 + ti * 16 + j + 8 * hi;
        int n = nb + n0 + tj * 16 + lr;
        float v = acc[ti][tj][j] + bias[n];
        size_t idx = (size_t)m * N + n;
        if (MODE == 0) {
          outb[idx] = (bf16_t)(v * cscale);
        } else if (MODE == 1) {
          outf[idx] = v + resid[idx];
        } else if (MODE == 2) {
          outb[idx] = (bf16_t)fmaxf(v, 0.0f);
        } else {
          outf[idx] = v;
        }
      }
}

// ---- Flash attention: per (b,h), 128 q-rows per workgroup ------------------
// Q/K/V are bf16 in [M][D] layout (head h occupies cols h*64..h*64+63).
// Q already carries the 1/sqrt(D) scale.

__global__ __launch_bounds__(256) void attn_kernel(
    const bf16_t* __restrict__ Qb, const bf16_t* __restrict__ Kb,
    const bf16_t* __restrict__ Vb, bf16_t* __restrict__ Ob) {
  __shared__ bf16_t Kt[64][72];        // K tile transposed: [hd][key]
  __shared__ bf16_t Vs[64][72];        // V tile natural:    [key][hd]
  __shared__ bf16_t Ps[8][16][72];     // per-wave P (q x key) bf16

  const int tid  = threadIdx.x;
  const int lane = tid & 31;
  const int wid  = tid >> 5;
  const int lr   = lane & 15;
  const int hi   = lane >> 4;
  const int bh   = blockIdx.y;
  const int b    = bh / kH;
  const int h    = bh % kH;
  const size_t rowbase = (size_t)b * kS;
  const int hoff = h * kHD;
  const int qrow = blockIdx.x * 128 + wid * 16 + lr;

  // Q fragments for this wave's 16 rows (k = 0..31 and 32..63 of head dim)
  const bf16_t* qptr = Qb + (rowbase + qrow) * kD + hoff;
  v16bf qf0 = fragA_ld(qptr + 0  + hi * 8);
  v16bf qf1 = fragA_ld(qptr + 32 + hi * 8);

  float mst[8], lst[8];
  v8f oacc[4];
#pragma unroll
  for (int j = 0; j < 8; ++j) { mst[j] = -3.0e38f; lst[j] = 0.0f; }
#pragma unroll
  for (int t = 0; t < 4; ++t) oacc[t] = v8f_zero();

  const int srow = tid >> 2;           // 0..63 (key within tile)
  const int scol = (tid & 3) * 16;     // 0,16,32,48 (head-dim segment)

  for (int kt = 0; kt < kS; kt += 64) {
    __syncthreads();
    // stage V natural (async path when available)
    {
      const bf16_t* src = Vb + (rowbase + kt + srow) * kD + hoff + scol;
#if HAVE_ASYNC_LDS
      async_copy_b128(&Vs[srow][scol + 0], src + 0);
      async_copy_b128(&Vs[srow][scol + 8], src + 8);
#else
      const uint4* s4 = reinterpret_cast<const uint4*>(src);
      uint4* d4 = reinterpret_cast<uint4*>(&Vs[srow][scol]);
      d4[0] = s4[0]; d4[1] = s4[1];
#endif
    }
    // stage K transposed: read [key][hd], write Kt[hd][key]
    {
      const bf16_t* src = Kb + (rowbase + kt + srow) * kD + hoff + scol;
      bf16_t tmp[16];
      *reinterpret_cast<uint4*>(tmp)     = *reinterpret_cast<const uint4*>(src);
      *reinterpret_cast<uint4*>(tmp + 8) = *reinterpret_cast<const uint4*>(src + 8);
#pragma unroll
      for (int i = 0; i < 16; ++i) Kt[scol + i][srow] = tmp[i];
    }
#if HAVE_ASYNC_LDS
    async_wait0();
#endif
    __syncthreads();

    // scores S = Q * K^T  (16 x 64 per wave); scale already folded into Q
    v8f s[4];
#pragma unroll
    for (int nt = 0; nt < 4; ++nt) {
      v8f z = v8f_zero();
      z = wmma_bf16(qf0, fragB_ld(&Kt[0  + hi * 8][nt * 16 + lr]), z);
      z = wmma_bf16(qf1, fragB_ld(&Kt[32 + hi * 8][nt * 16 + lr]), z);
      s[nt] = z;
    }

    // online softmax (row reductions across the 16-lane half-groups)
    float tmax[8];
#pragma unroll
    for (int j = 0; j < 8; ++j) {
      float t = fmaxf(fmaxf(s[0][j], s[1][j]), fmaxf(s[2][j], s[3][j]));
#pragma unroll
      for (int msk = 1; msk < 16; msk <<= 1) t = fmaxf(t, __shfl_xor(t, msk));
      tmax[j] = t;
    }
#pragma unroll
    for (int j = 0; j < 8; ++j) {
      float mn = fmaxf(mst[j], tmax[j]);
      float es = __expf(mst[j] - mn);
      mst[j] = mn;
      float rs = 0.0f;
#pragma unroll
      for (int nt = 0; nt < 4; ++nt) {
        float p = __expf(s[nt][j] - mn);
        s[nt][j] = p;
        rs += p;
      }
#pragma unroll
      for (int msk = 1; msk < 16; msk <<= 1) rs += __shfl_xor(rs, msk);
      lst[j] = lst[j] * es + rs;
#pragma unroll
      for (int nt = 0; nt < 4; ++nt) oacc[nt][j] *= es;
    }

    // P (C-layout) -> LDS -> A-layout fragments
#pragma unroll
    for (int nt = 0; nt < 4; ++nt)
#pragma unroll
      for (int j = 0; j < 8; ++j)
        Ps[wid][j + 8 * hi][nt * 16 + lr] = (bf16_t)s[nt][j];
    __syncthreads();

    v16bf pa0 = fragA_ld(&Ps[wid][lr][0  + hi * 8]);
    v16bf pa1 = fragA_ld(&Ps[wid][lr][32 + hi * 8]);
#pragma unroll
    for (int nt = 0; nt < 4; ++nt) {
      oacc[nt] = wmma_bf16(pa0, fragB_ld(&Vs[0  + hi * 8][nt * 16 + lr]), oacc[nt]);
      oacc[nt] = wmma_bf16(pa1, fragB_ld(&Vs[32 + hi * 8][nt * 16 + lr]), oacc[nt]);
    }
  }

  // normalize and store concat-head output (bf16, [M][D])
#pragma unroll
  for (int nt = 0; nt < 4; ++nt)
#pragma unroll
    for (int j = 0; j < 8; ++j) {
      int m = blockIdx.x * 128 + wid * 16 + j + 8 * hi;
      float v = oacc[nt][j] / lst[j];
      Ob[(rowbase + m) * kD + hoff + nt * 16 + lr] = (bf16_t)v;
    }
}

// ---- LayerNorm over D=512 (one row per workgroup) --------------------------

__global__ __launch_bounds__(256) void ln_kernel(
    const float* __restrict__ res, const float* __restrict__ gamma,
    const float* __restrict__ beta, bf16_t* __restrict__ out) {
  const int row = blockIdx.x;
  const int tid = threadIdx.x;
  const float* r = res + (size_t)row * kD;
  float x0 = r[tid], x1 = r[tid + 256];
  float s = x0 + x1, q = x0 * x0 + x1 * x1;
#pragma unroll
  for (int m = 1; m < 32; m <<= 1) { s += __shfl_xor(s, m); q += __shfl_xor(q, m); }
  __shared__ float sred[8], qred[8];
  if ((tid & 31) == 0) { sred[tid >> 5] = s; qred[tid >> 5] = q; }
  __syncthreads();
  float ts = 0.0f, tq = 0.0f;
#pragma unroll
  for (int i = 0; i < 8; ++i) { ts += sred[i]; tq += qred[i]; }
  float mu   = ts * (1.0f / kD);
  float var  = tq * (1.0f / kD) - mu * mu;
  float rstd = rsqrtf(var + 1e-5f);
  out[(size_t)row * kD + tid]       = (bf16_t)((x0 - mu) * rstd * gamma[tid] + beta[tid]);
  out[(size_t)row * kD + tid + 256] = (bf16_t)((x1 - mu) * rstd * gamma[tid + 256] + beta[tid + 256]);
}

// ---- mask passthrough (reference returns (out, mask)) ----------------------

__global__ void mask_kernel(const unsigned char* __restrict__ m,
                            float* __restrict__ out, int n) {
  int i = blockIdx.x * 256 + threadIdx.x;
  if (i < n) out[i] = m[i] ? 1.0f : 0.0f;
}

// ---------------------------------------------------------------------------

extern "C" void kernel_launch(void* const* d_in, const int* in_sizes, int n_in,
                              void* d_out, int out_size, void* d_ws, size_t ws_size,
                              hipStream_t stream) {
  (void)in_sizes; (void)n_in; (void)ws_size;

  const float* emb  = (const float*)d_in[0];
  const unsigned char* mask = (const unsigned char*)d_in[1];
  const float* Wq   = (const float*)d_in[2];
  const float* bq   = (const float*)d_in[3];
  const float* Wk   = (const float*)d_in[4];
  const float* bk   = (const float*)d_in[5];
  const float* Wv   = (const float*)d_in[6];
  const float* bv   = (const float*)d_in[7];
  const float* Wc   = (const float*)d_in[8];
  const float* bc   = (const float*)d_in[9];
  const float* W1   = (const float*)d_in[10];
  const float* b1   = (const float*)d_in[11];
  const float* W2   = (const float*)d_in[12];
  const float* b2   = (const float*)d_in[13];
  const float* gamma = (const float*)d_in[14];
  const float* beta  = (const float*)d_in[15];

  char* ws = (char*)d_ws;
  size_t off = 0;
  auto take = [&](size_t bytes) -> char* {
    char* p = ws + off;
    off += (bytes + 255) & ~(size_t)255;
    return p;
  };

  bf16_t* embB = (bf16_t*)take((size_t)kM * kD * 2);
  bf16_t* WqB  = (bf16_t*)take((size_t)kH * kD * kHD * 2);
  bf16_t* WkB  = (bf16_t*)take((size_t)kH * kD * kHD * 2);
  bf16_t* WvB  = (bf16_t*)take((size_t)kH * kD * kHD * 2);
  bf16_t* WcB  = (bf16_t*)take((size_t)kD * kD * 2);
  bf16_t* W1B  = (bf16_t*)take((size_t)kD * kFF * 2);
  bf16_t* W2B  = (bf16_t*)take((size_t)kFF * kD * 2);
  bf16_t* Qb   = (bf16_t*)take((size_t)kM * kD * 2);
  bf16_t* Kb   = (bf16_t*)take((size_t)kM * kD * 2);
  bf16_t* Vb   = (bf16_t*)take((size_t)kM * kD * 2);
  bf16_t* Ob   = (bf16_t*)take((size_t)kM * kD * 2);
  float*  resf = (float*)take((size_t)kM * kD * 4);
  bf16_t* Hnb  = (bf16_t*)take((size_t)kM * kD * 2);
  bf16_t* Gb   = (bf16_t*)take((size_t)kM * kFF * 2);

  auto cvt = [&](const float* in, bf16_t* outp, int n) {
    cvt_kernel<<<(n + 255) / 256, 256, 0, stream>>>(in, outp, n);
  };

  // 1) bf16 converts
  cvt(emb, embB, kM * kD);
  cvt(Wq, WqB, kH * kD * kHD);
  cvt(Wk, WkB, kH * kD * kHD);
  cvt(Wv, WvB, kH * kD * kHD);
  cvt(Wc, WcB, kD * kD);
  cvt(W1, W1B, kD * kFF);
  cvt(W2, W2B, kFF * kD);

  // 2) QKV projections: [8192 x 512] x per-head [512 x 64] blocks.
  //    Q gets the 1/sqrt(D) attention scale folded into its epilogue.
  {
    dim3 grid(kD / 64, kM / 128);
    gemm_bf16_kernel<0><<<grid, 256, 0, stream>>>(
        embB, WqB, bq, nullptr, Qb, nullptr, kM, kD, kD, kHD, (long)kD * kHD,
        kInvScale);
    gemm_bf16_kernel<0><<<grid, 256, 0, stream>>>(
        embB, WkB, bk, nullptr, Kb, nullptr, kM, kD, kD, kHD, (long)kD * kHD,
        1.0f);
    gemm_bf16_kernel<0><<<grid, 256, 0, stream>>>(
        embB, WvB, bv, nullptr, Vb, nullptr, kM, kD, kD, kHD, (long)kD * kHD,
        1.0f);
  }

  // 3) flash attention
  {
    dim3 grid(kS / 128, kB * kH);
    attn_kernel<<<grid, 256, 0, stream>>>(Qb, Kb, Vb, Ob);
  }

  // 4) output projection + residual (f32)
  {
    dim3 grid(kD / 64, kM / 128);
    gemm_bf16_kernel<1><<<grid, 256, 0, stream>>>(
        Ob, WcB, bc, emb, nullptr, resf, kM, kD, kD, kD, 64L, 1.0f);
  }

  // 5) LayerNorm
  ln_kernel<<<kM, 256, 0, stream>>>(resf, gamma, beta, Hnb);

  // 6) FFN1 + ReLU
  {
    dim3 grid(kFF / 64, kM / 128);
    gemm_bf16_kernel<2><<<grid, 256, 0, stream>>>(
        Hnb, W1B, b1, nullptr, Gb, nullptr, kM, kFF, kD, kFF, 64L, 1.0f);
  }

  // 7) FFN2 -> d_out (f32)
  {
    dim3 grid(kD / 64, kM / 128);
    gemm_bf16_kernel<3><<<grid, 256, 0, stream>>>(
        Gb, W2B, b2, nullptr, nullptr, (float*)d_out, kM, kD, kFF, kD, 64L,
        1.0f);
  }

  // 8) mask passthrough (second tuple element), if the harness sized it in
  if (out_size >= kM * kD + kB * kS) {
    float* mout = (float*)d_out + (size_t)kM * kD;
    mask_kernel<<<(kB * kS + 255) / 256, 256, 0, stream>>>(mask, mout, kB * kS);
  }
}
